// DynamicGraphTemporalModel_85770496901317
// MI455X (gfx1250) — matrix operand order repaired and checked by reading, not compiled, not run on hardware
//
#include <hip/hip_runtime.h>

typedef __attribute__((ext_vector_type(16))) _Float16 v16h;
typedef __attribute__((ext_vector_type(8)))  _Float16 v8h;
typedef __attribute__((ext_vector_type(8)))  float    v8f;

#define N_NODES 19
#define NPAD    32
#define GEDIM   64
#define T_LEN   256
#define BT_TOT  (256 * 256)

// ---------------------------------------------------------------------------
// WMMA helpers (gfx1250, wave32): D(16x16,f32) = A(16x32,f16) x B(32x16,f16) + C
// ---------------------------------------------------------------------------
__device__ __forceinline__ v8f wmma_f16(v16h a, v16h b, v8f c) {
  // 8 args: (neg_a, A, neg_b, B, c_mod, C, reuse_a, reuse_b)
  return __builtin_amdgcn_wmma_f32_16x16x32_f16(false, a, false, b, (short)0, c,
                                                false, false);
}

// A-fragment: 16x32 tile from row-major [rows x ld] f16, tile rows = row0..row0+15.
// Lane (ln = lane&15, half = lane>>4): elems 0..7 = K[8*half .. 8*half+7],
// elems 8..15 = K[16+8*half .. 23+8*half]  (ISA 16-bit A 16x32 layout).
__device__ __forceinline__ v16h frag_a(const _Float16* base, int ld, int row0,
                                       int k0, int ln, int half) {
  const _Float16* p = base + (row0 + ln) * ld + k0 + 8 * half;
  v8h lo = *(const v8h*)(p);
  v8h hi = *(const v8h*)(p + 16);
  return __builtin_shufflevector(lo, hi, 0, 1, 2, 3, 4, 5, 6, 7,
                                 8, 9, 10, 11, 12, 13, 14, 15);
}

// B-fragment: B = W^T where W is row-major [N x ld] f16; tile cols = n0..n0+15.
// Lane (ln, half): elems 0..7 = W[n0+ln][k0+8*half .. +7], elems 8..15 = W[n0+ln][k0+16+8*half .. +7].
__device__ __forceinline__ v16h frag_b(const _Float16* base, int ld, int n0,
                                       int k0, int ln, int half) {
  const _Float16* p = base + (n0 + ln) * ld + k0 + 8 * half;
  v8h lo = *(const v8h*)(p);
  v8h hi = *(const v8h*)(p + 16);
  return __builtin_shufflevector(lo, hi, 0, 1, 2, 3, 4, 5, 6, 7,
                                 8, 9, 10, 11, 12, 13, 14, 15);
}

__device__ __forceinline__ float sigmf(float x) {
  return 1.0f / (1.0f + expf(-x));
}

// ---------------------------------------------------------------------------
// Prep: fp32 weights -> f16 (w1 K-padded 19->32), combined LSTM biases.
// ---------------------------------------------------------------------------
__global__ void prep_kernel(
    const float* __restrict__ w1_w, const float* __restrict__ w2_w,
    const float* __restrict__ Wih0f, const float* __restrict__ Whh0f,
    const float* __restrict__ bih0, const float* __restrict__ bhh0,
    const float* __restrict__ Wih1f, const float* __restrict__ Whh1f,
    const float* __restrict__ bih1, const float* __restrict__ bhh1,
    _Float16* __restrict__ w1p, _Float16* __restrict__ w2p,
    _Float16* __restrict__ Wih0, _Float16* __restrict__ Whh0,
    _Float16* __restrict__ Wih1, _Float16* __restrict__ Whh1,
    float* __restrict__ bias0, float* __restrict__ bias1) {
  const int i0 = blockIdx.x * blockDim.x + threadIdx.x;
  const int st = gridDim.x * blockDim.x;
  for (int x = i0; x < 64 * NPAD; x += st) {
    int r = x >> 5, c = x & 31;
    w1p[x] = (_Float16)((c < N_NODES) ? w1_w[r * N_NODES + c] : 0.0f);
  }
  for (int x = i0; x < 64 * 64; x += st) w2p[x] = (_Float16)w2_w[x];
  for (int x = i0; x < 256 * 64; x += st) {
    Wih0[x] = (_Float16)Wih0f[x];
    Whh0[x] = (_Float16)Whh0f[x];
    Wih1[x] = (_Float16)Wih1f[x];
    Whh1[x] = (_Float16)Whh1f[x];
  }
  for (int x = i0; x < 256; x += st) {
    bias0[x] = bih0[x] + bhh0[x];
    bias1[x] = bih1[x] + bhh1[x];
  }
}

// ---------------------------------------------------------------------------
// Graph encoder: one workgroup (256 threads = 8 waves) per (b,t).
//   An = norm-adj(A) padded to 32x32 f16 (zeros outside 19x19)
//   Y1 = A@w1^T + b1            (WMMA, stored transposed -> B operand of GEMM2)
//   X1 = relu(An@Y1)            (WMMA)
//   Y2 = X1@w2^T + b2           (WMMA, stored transposed)
//   X2 = relu(An@Y2); emb = mean over 19 node rows, * mask
// ---------------------------------------------------------------------------
__global__ __launch_bounds__(256) void graph_enc_kernel(
    const float* __restrict__ conn, const int* __restrict__ mask,
    const float* __restrict__ w1_b, const float* __restrict__ w2_b,
    const _Float16* __restrict__ w1p, const _Float16* __restrict__ w2p,
    _Float16* __restrict__ emb) {
  __shared__ float lA[N_NODES * N_NODES];
  __shared__ float ldis[N_NODES];
  __shared__ __align__(32) _Float16 Apad[NPAD * NPAD];
  __shared__ __align__(32) _Float16 Anp[NPAD * NPAD];
  __shared__ __align__(32) _Float16 S1t[64 * NPAD];   // Y1^T  (N x K)
  __shared__ __align__(32) _Float16 X1[NPAD * 64];    // row-major (M x K)
  __shared__ __align__(32) _Float16 Y2t[64 * NPAD];   // Y2^T  (N x K)
  __shared__ float lpart[4][64];

  const int bt = blockIdx.x;
  const int tid = threadIdx.x;
  const float* A = conn + (size_t)bt * (N_NODES * N_NODES);

  for (int i = tid; i < N_NODES * N_NODES; i += 256) lA[i] = A[i];
  __syncthreads();
  if (tid < N_NODES) {
    float s = 1.0f;  // self loop
    for (int j = 0; j < N_NODES; ++j) s += lA[tid * N_NODES + j];
    if (s < 1e-6f) s = 1e-6f;
    ldis[tid] = rsqrtf(s);
  }
  __syncthreads();
  for (int i = tid; i < NPAD * NPAD; i += 256) {
    int r = i >> 5, c = i & 31;
    float a = 0.0f, an = 0.0f;
    if (r < N_NODES && c < N_NODES) {
      a = lA[r * N_NODES + c];
      an = (a + (r == c ? 1.0f : 0.0f)) * ldis[r] * ldis[c];
    }
    Apad[i] = (_Float16)a;
    Anp[i] = (_Float16)an;
  }
  __syncthreads();

  const int wave = tid >> 5, lane = tid & 31;
  const int half = lane >> 4, ln = lane & 15;
  const int mt = wave >> 2, nt = wave & 3;  // 2 x 4 output tiles (32 x 64)

  {  // GEMM1: Y1 = Apad @ w1p^T + b1  -> S1t (transposed, f16)
    v8f c = {};
    c = wmma_f16(frag_a(Apad, NPAD, mt * 16, 0, ln, half),
                 frag_b(w1p, NPAD, nt * 16, 0, ln, half), c);
    float bn = w1_b[nt * 16 + ln];
    _Float16* dst = &S1t[(nt * 16 + ln) * NPAD + mt * 16 + 8 * half];
#pragma unroll
    for (int r = 0; r < 8; ++r) dst[r] = (_Float16)(c[r] + bn);
  }
  __syncthreads();
  {  // GEMM2: X1 = relu(Anp @ Y1)
    v8f c = {};
    c = wmma_f16(frag_a(Anp, NPAD, mt * 16, 0, ln, half),
                 frag_b(S1t, NPAD, nt * 16, 0, ln, half), c);
#pragma unroll
    for (int r = 0; r < 8; ++r) {
      int m = mt * 16 + r + 8 * half;
      X1[m * 64 + nt * 16 + ln] = (_Float16)fmaxf(c[r], 0.0f);
    }
  }
  __syncthreads();
  {  // GEMM3: Y2 = X1 @ w2p^T + b2  -> Y2t (transposed, f16)
    v8f c = {};
    c = wmma_f16(frag_a(X1, 64, mt * 16, 0, ln, half),
                 frag_b(w2p, 64, nt * 16, 0, ln, half), c);
    c = wmma_f16(frag_a(X1, 64, mt * 16, 32, ln, half),
                 frag_b(w2p, 64, nt * 16, 32, ln, half), c);
    float bn = w2_b[nt * 16 + ln];
    _Float16* dst = &Y2t[(nt * 16 + ln) * NPAD + mt * 16 + 8 * half];
#pragma unroll
    for (int r = 0; r < 8; ++r) dst[r] = (_Float16)(c[r] + bn);
  }
  __syncthreads();
  {  // GEMM4: X2 = relu(Anp @ Y2); deterministic column partial sums (rows<19)
    v8f c = {};
    c = wmma_f16(frag_a(Anp, NPAD, mt * 16, 0, ln, half),
                 frag_b(Y2t, NPAD, nt * 16, 0, ln, half), c);
    float s = 0.0f;
#pragma unroll
    for (int r = 0; r < 8; ++r) {
      int m = mt * 16 + r + 8 * half;
      if (m < N_NODES) s += fmaxf(c[r], 0.0f);
    }
    lpart[mt * 2 + half][nt * 16 + ln] = s;
  }
  __syncthreads();
  if (tid < 64) {
    float s = lpart[0][tid] + lpart[1][tid] + lpart[2][tid] + lpart[3][tid];
    float mf = (float)mask[bt];
    emb[(size_t)bt * GEDIM + tid] = (_Float16)(s * (1.0f / 19.0f) * mf);
  }
}

// ---------------------------------------------------------------------------
// Fused 2-layer LSTM + FC head. 16 blocks x 16 batch rows; 8 waves per block.
// Per step/layer: gates(16x256) = x@Wih^T + h@Whh^T + b  (4 WMMAs per 16x16
// gate tile; 16 N-tiles spread 2-per-wave), then pointwise state update.
// ---------------------------------------------------------------------------
__global__ __launch_bounds__(256) void lstm_head_kernel(
    const _Float16* __restrict__ emb, const int* __restrict__ mask,
    const _Float16* __restrict__ Wih0, const _Float16* __restrict__ Whh0,
    const float* __restrict__ bias0,
    const _Float16* __restrict__ Wih1, const _Float16* __restrict__ Whh1,
    const float* __restrict__ bias1,
    const float* __restrict__ fc1_w, const float* __restrict__ fc1_b,
    const float* __restrict__ fc2_w, const float* __restrict__ fc2_b,
    float* __restrict__ out) {
  __shared__ __align__(32) _Float16 x0[16 * 64];
  __shared__ __align__(32) _Float16 h0[16 * 64];
  __shared__ __align__(32) _Float16 h1[16 * 64];
  __shared__ float c0[16 * 64];
  __shared__ float c1[16 * 64];
  __shared__ float gates[16 * 256];
  __shared__ float lastH[16 * 64];
  __shared__ float hfc[16 * 32];
  __shared__ int lastidx[16];

  const int bb = blockIdx.x * 16;
  const int tid = threadIdx.x;
  const int wave = tid >> 5, lane = tid & 31;
  const int half = lane >> 4, ln = lane & 15;

  for (int i = tid; i < 16 * 64; i += 256) {
    h0[i] = (_Float16)0.0f;
    h1[i] = (_Float16)0.0f;
    c0[i] = 0.0f;
    c1[i] = 0.0f;
    lastH[i] = 0.0f;
  }
  if (tid < 16) {
    int len = 0;
    for (int t = 0; t < T_LEN; ++t) len += mask[(bb + tid) * T_LEN + t];
    if (len < 1) len = 1;
    lastidx[tid] = len - 1;
  }
  __syncthreads();

  for (int t = 0; t < T_LEN; ++t) {
    for (int i = tid; i < 16 * 64; i += 256) {
      int r = i >> 6, k = i & 63;
      x0[i] = emb[(((size_t)(bb + r)) * T_LEN + t) * GEDIM + k];
    }
    __syncthreads();
    // ---- layer 0 gates ----
#pragma unroll
    for (int s = 0; s < 2; ++s) {
      const int n0 = (wave * 2 + s) * 16;
      v8f c = {};
      c = wmma_f16(frag_a(x0, 64, 0, 0, ln, half),
                   frag_b(Wih0, 64, n0, 0, ln, half), c);
      c = wmma_f16(frag_a(x0, 64, 0, 32, ln, half),
                   frag_b(Wih0, 64, n0, 32, ln, half), c);
      c = wmma_f16(frag_a(h0, 64, 0, 0, ln, half),
                   frag_b(Whh0, 64, n0, 0, ln, half), c);
      c = wmma_f16(frag_a(h0, 64, 0, 32, ln, half),
                   frag_b(Whh0, 64, n0, 32, ln, half), c);
      float bn = bias0[n0 + ln];
#pragma unroll
      for (int r = 0; r < 8; ++r)
        gates[(r + 8 * half) * 256 + n0 + ln] = c[r] + bn;
    }
    __syncthreads();
    for (int i = tid; i < 16 * 64; i += 256) {
      int r = i >> 6, hh = i & 63;
      const float* g = &gates[r * 256];
      float cn = sigmf(g[64 + hh]) * c0[i] + sigmf(g[hh]) * tanhf(g[128 + hh]);
      c0[i] = cn;
      h0[i] = (_Float16)(sigmf(g[192 + hh]) * tanhf(cn));
    }
    __syncthreads();
    // ---- layer 1 gates (input = new h0) ----
#pragma unroll
    for (int s = 0; s < 2; ++s) {
      const int n0 = (wave * 2 + s) * 16;
      v8f c = {};
      c = wmma_f16(frag_a(h0, 64, 0, 0, ln, half),
                   frag_b(Wih1, 64, n0, 0, ln, half), c);
      c = wmma_f16(frag_a(h0, 64, 0, 32, ln, half),
                   frag_b(Wih1, 64, n0, 32, ln, half), c);
      c = wmma_f16(frag_a(h1, 64, 0, 0, ln, half),
                   frag_b(Whh1, 64, n0, 0, ln, half), c);
      c = wmma_f16(frag_a(h1, 64, 0, 32, ln, half),
                   frag_b(Whh1, 64, n0, 32, ln, half), c);
      float bn = bias1[n0 + ln];
#pragma unroll
      for (int r = 0; r < 8; ++r)
        gates[(r + 8 * half) * 256 + n0 + ln] = c[r] + bn;
    }
    __syncthreads();
    for (int i = tid; i < 16 * 64; i += 256) {
      int r = i >> 6, hh = i & 63;
      const float* g = &gates[r * 256];
      float cn = sigmf(g[64 + hh]) * c1[i] + sigmf(g[hh]) * tanhf(g[128 + hh]);
      c1[i] = cn;
      float hn = sigmf(g[192 + hh]) * tanhf(cn);
      h1[i] = (_Float16)hn;
      if (t == lastidx[r]) lastH[i] = hn;
    }
    __syncthreads();
  }

  // ---- FC head (fp32 VALU) ----
  for (int i = tid; i < 16 * 32; i += 256) {
    int r = i >> 5, j = i & 31;
    float s = fc1_b[j];
    for (int k = 0; k < 64; ++k) s += lastH[r * 64 + k] * fc1_w[j * 64 + k];
    hfc[i] = fmaxf(s, 0.0f);
  }
  __syncthreads();
  if (tid < 32) {
    int r = tid >> 1, c = tid & 1;
    float s = fc2_b[c];
    for (int j = 0; j < 32; ++j) s += hfc[r * 32 + j] * fc2_w[c * 32 + j];
    out[(bb + r) * 2 + c] = s;
  }
}

// ---------------------------------------------------------------------------
extern "C" void kernel_launch(void* const* d_in, const int* in_sizes, int n_in,
                              void* d_out, int out_size, void* d_ws,
                              size_t ws_size, hipStream_t stream) {
  (void)in_sizes; (void)n_in; (void)out_size; (void)ws_size;

  const float* conn = (const float*)d_in[0];
  const int* mask = (const int*)d_in[1];
  const float* w1_w = (const float*)d_in[2];
  const float* w1_b = (const float*)d_in[3];
  const float* w2_w = (const float*)d_in[4];
  const float* w2_b = (const float*)d_in[5];
  const float* Wih0f = (const float*)d_in[6];
  const float* Whh0f = (const float*)d_in[7];
  const float* bih0 = (const float*)d_in[8];
  const float* bhh0 = (const float*)d_in[9];
  const float* Wih1f = (const float*)d_in[10];
  const float* Whh1f = (const float*)d_in[11];
  const float* bih1 = (const float*)d_in[12];
  const float* bhh1 = (const float*)d_in[13];
  const float* fc1_w = (const float*)d_in[14];
  const float* fc1_b = (const float*)d_in[15];
  const float* fc2_w = (const float*)d_in[16];
  const float* fc2_b = (const float*)d_in[17];

  char* ws = (char*)d_ws;
  _Float16* w1p = (_Float16*)(ws + 0);           //  4 KB  (64 x 32 f16)
  _Float16* w2p = (_Float16*)(ws + 4096);        //  8 KB  (64 x 64 f16)
  _Float16* Wih0 = (_Float16*)(ws + 12288);      // 32 KB  (256 x 64 f16)
  _Float16* Whh0 = (_Float16*)(ws + 45056);      // 32 KB
  _Float16* Wih1 = (_Float16*)(ws + 77824);      // 32 KB
  _Float16* Whh1 = (_Float16*)(ws + 110592);     // 32 KB
  float* bias0 = (float*)(ws + 143360);          //  1 KB
  float* bias1 = (float*)(ws + 144384);          //  1 KB
  _Float16* emb = (_Float16*)(ws + 145408);      //  8 MB  (B*T x 64 f16)

  prep_kernel<<<dim3(64), dim3(256), 0, stream>>>(
      w1_w, w2_w, Wih0f, Whh0f, bih0, bhh0, Wih1f, Whh1f, bih1, bhh1, w1p, w2p,
      Wih0, Whh0, Wih1, Whh1, bias0, bias1);

  graph_enc_kernel<<<dim3(BT_TOT), dim3(256), 0, stream>>>(
      conn, mask, w1_b, w2_b, w1p, w2p, emb);

  lstm_head_kernel<<<dim3(16), dim3(256), 0, stream>>>(
      emb, mask, Wih0, Whh0, bias0, Wih1, Whh1, bias1, fc1_w, fc1_b, fc2_w,
      fc2_b, (float*)d_out);
}
